// Hercules_47467978556114
// MI455X (gfx1250) — matrix operand order repaired and checked by reading, not compiled, not run on hardware
//
#include <hip/hip_runtime.h>

typedef __attribute__((ext_vector_type(16))) _Float16 v16h;
typedef __attribute__((ext_vector_type(8)))  _Float16 v8h;
typedef __attribute__((ext_vector_type(4)))  _Float16 v4h;
typedef __attribute__((ext_vector_type(8)))  float    v8f;
typedef __attribute__((ext_vector_type(4)))  float    v4f;

#define DIM  1024
#define MTOT 32768          // 4 batches * 8192 seq

union V16 { v16h v; v8h h[2]; };

// CDNA5 async DMA: global -> LDS, 16B per lane, tracked by ASYNCcnt.
__device__ __forceinline__ void async_load_b128(const void* gptr, void* lptr) {
  unsigned loff = (unsigned)(unsigned long long)lptr;   // low 32 bits = LDS offset
  asm volatile("global_load_async_to_lds_b128 %0, %1, off"
               :: "v"(loff), "v"(gptr));
}
__device__ __forceinline__ void wait_async0() {
  asm volatile("s_wait_asynccnt 0x0" ::: "memory");
}

// ---------------- prep: f32 -> f16 convert (X) ----------------
__global__ __launch_bounds__(256) void cvt_x(const float* __restrict__ X,
                                             _Float16* __restrict__ Xh, int n4) {
  int i = blockIdx.x * 256 + threadIdx.x;
  if (i >= n4) return;
  v4f x = *(const v4f*)(X + (size_t)i * 4);
  v4h h;
  h[0] = (_Float16)x[0]; h[1] = (_Float16)x[1];
  h[2] = (_Float16)x[2]; h[3] = (_Float16)x[3];
  *(v4h*)(Xh + (size_t)i * 4) = h;
}

// ---------------- prep: f32 [R][C] -> f16 [C][R] (transpose+convert) ----------------
__global__ __launch_bounds__(256) void transpose_cvt(const float* __restrict__ src,
                                                     _Float16* __restrict__ dst,
                                                     int R, int C) {
  __shared__ float tile[32][33];
  int x  = blockIdx.x * 32 + threadIdx.x;   // source col
  int y0 = blockIdx.y * 32;                 // source row base
  for (int j = threadIdx.y; j < 32; j += 8)
    tile[j][threadIdx.x] = src[(size_t)(y0 + j) * C + x];
  __syncthreads();
  int r = y0 + threadIdx.x;                 // dest inner index (coalesced)
  for (int j = threadIdx.y; j < 32; j += 8)
    dst[(size_t)(blockIdx.x * 32 + j) * R + r] = (_Float16)tile[threadIdx.x][j];
}

// ---------------- GEMM1: qkv = Xh @ Wi + bi; store Qh; accumulate gate A ----------------
// Tile 128(M) x 64(N), Q/K/V share one A tile. Double-buffered LDS, async DMA staging,
// K-loop manually unrolled x2 so buffer addresses are compile-time constants.
#define G1_A (128 * 32)
#define G1_B (3 * 64 * 32)
__global__ __launch_bounds__(256) void gemm1_qkv(const _Float16* __restrict__ Xh,   // [32768][1024]
                                                 const _Float16* __restrict__ WiT,  // [3072][1024]
                                                 const float* __restrict__ bi,      // [3072]
                                                 _Float16* __restrict__ Qh,         // [32768][1024]
                                                 float* __restrict__ Aacc)          // [4][1024]
{
  __shared__ __align__(32) _Float16 ldsA[2 * G1_A];
  __shared__ __align__(32) _Float16 ldsB[2 * G1_B];

  const int tid  = threadIdx.x;
  const int w    = tid >> 5, lane = tid & 31;
  const int wm   = w & 3,    wn   = w >> 2;
  const int m_base = blockIdx.y * 128;
  const int n_base = blockIdx.x * 64;
  const int batch  = m_base >> 13;          // 8192 rows/batch; tiles never cross
  const int rr = lane & 15, hi = lane >> 4;
  const int klo = hi * 8;
  const int r = tid >> 2, part = tid & 3;   // r: 0..63, part: 0..3 (8 halfs each)

  v8f acc[3][2][2];
  v8f zero = {};
  for (int o = 0; o < 3; ++o)
    for (int a = 0; a < 2; ++a)
      for (int b = 0; b < 2; ++b) acc[o][a][b] = zero;

  auto issue = [&](int kc, _Float16* bufA, _Float16* bufB) {
    const int k0 = kc * 32;
#pragma unroll
    for (int rep = 0; rep < 2; ++rep) {
      int row = r + rep * 64;
      async_load_b128(Xh + (size_t)(m_base + row) * DIM + k0 + part * 8,
                      &bufA[row * 32 + part * 8]);
    }
#pragma unroll
    for (int o = 0; o < 3; ++o) {
      int e = n_base + r + o * DIM;
      async_load_b128(WiT + (size_t)e * DIM + k0 + part * 8,
                      &bufB[o * 2048 + r * 32 + part * 8]);
    }
  };

  auto compute = [&](const _Float16* bufA, const _Float16* bufB) {
    V16 af[2];
#pragma unroll
    for (int mt = 0; mt < 2; ++mt) {
      int row = wm * 32 + mt * 16 + rr;
      af[mt].h[0] = *(const v8h*)(&bufA[row * 32 + klo * 2]);       // K  0..7  / 16..23
      af[mt].h[1] = *(const v8h*)(&bufA[row * 32 + klo * 2 + 8]);  // K  8..15 / 24..31
    }
#pragma unroll
    for (int o = 0; o < 3; ++o) {
#pragma unroll
      for (int nt = 0; nt < 2; ++nt) {
        int col = wn * 32 + nt * 16 + rr;
        v16h bf = *(const v16h*)(&bufB[o * 2048 + col * 32 + hi * 16]);
#pragma unroll
        for (int mt = 0; mt < 2; ++mt) {
          acc[o][mt][nt] = __builtin_amdgcn_wmma_f32_16x16x32_f16(
              false, af[mt].v, false, bf, (short)0, acc[o][mt][nt], false, false);
        }
      }
    }
  };

  issue(0, ldsA, ldsB);
#pragma unroll 1
  for (int kc = 0; kc < 32; kc += 2) {
    wait_async0();
    __syncthreads();
    issue(kc + 1, ldsA + G1_A, ldsB + G1_B);   // kc+1 <= 31 always
    compute(ldsA, ldsB);

    wait_async0();
    __syncthreads();
    if (kc + 2 < 32) issue(kc + 2, ldsA, ldsB);
    compute(ldsA + G1_A, ldsB + G1_B);
  }

  // epilogue: Q store + relu(K)*relu(V) reduction
#pragma unroll
  for (int nt = 0; nt < 2; ++nt) {
    int n = n_base + wn * 32 + nt * 16 + rr;
    float bq = bi[n], bk = bi[DIM + n], bv = bi[2 * DIM + n];
    float gsum = 0.f;
#pragma unroll
    for (int mt = 0; mt < 2; ++mt) {
      int m0 = m_base + wm * 32 + mt * 16 + hi * 8;
#pragma unroll
      for (int j = 0; j < 8; ++j) {
        float q = acc[0][mt][nt][j] + bq;
        Qh[(size_t)(m0 + j) * DIM + n] = (_Float16)q;
        float kk = acc[1][mt][nt][j] + bk; kk = kk > 0.f ? kk : 0.f;
        float vv = acc[2][mt][nt][j] + bv; vv = vv > 0.f ? vv : 0.f;
        gsum += kk * vv;
      }
    }
    gsum += __shfl_xor(gsum, 16);
    if (hi == 0) atomicAdd(&Aacc[batch * DIM + n], gsum);
  }
}

// ---------------- finalize gate: Ah = f16(0.5*A + 0.5) ----------------
__global__ __launch_bounds__(256) void finalize_gate(const float* __restrict__ Aacc,
                                                     _Float16* __restrict__ Ah, int n) {
  int i = blockIdx.x * 256 + threadIdx.x;
  if (i < n) Ah[i] = (_Float16)(0.5f * Aacc[i] + 0.5f);
}

// ---------------- GEMM2: out = (Ah .* Qh) @ Wo + bo + X ----------------
// Tile 128(M) x 128(N). Double-buffered, manually unrolled x2; A tile via register
// path (gate mul), B tile via async DMA.
#define G2_T (128 * 32)
__global__ __launch_bounds__(256) void gemm2_out(const _Float16* __restrict__ Qh,   // [32768][1024]
                                                 const _Float16* __restrict__ Ah,   // [4][1024]
                                                 const _Float16* __restrict__ WoT,  // [1024][1024]
                                                 const float* __restrict__ bo,      // [1024]
                                                 const float* __restrict__ X,       // [32768][1024]
                                                 float* __restrict__ out)           // [32768][1024]
{
  __shared__ __align__(32) _Float16 ldsA[2 * G2_T];
  __shared__ __align__(32) _Float16 ldsB[2 * G2_T];

  const int tid  = threadIdx.x;
  const int w    = tid >> 5, lane = tid & 31;
  const int wm   = w & 3,    wn   = w >> 2;
  const int m_base = blockIdx.y * 128;
  const int n_base = blockIdx.x * 128;
  const int batch  = m_base >> 13;
  const int rr = lane & 15, hi = lane >> 4;
  const int klo = hi * 8;
  const int r = tid >> 2, part = tid & 3;

  v8f acc[2][4];
  v8f zero = {};
  for (int a = 0; a < 2; ++a)
    for (int b = 0; b < 4; ++b) acc[a][b] = zero;

  auto stage = [&](int kc, _Float16* bufA, _Float16* bufB) {
    const int k0 = kc * 32;
#pragma unroll
    for (int rep = 0; rep < 2; ++rep) {
      int row = r + rep * 64;
      async_load_b128(WoT + (size_t)(n_base + row) * DIM + k0 + part * 8,
                      &bufB[row * 32 + part * 8]);
    }
    v8h g = *(const v8h*)(Ah + batch * DIM + k0 + part * 8);   // gate along K dim
#pragma unroll
    for (int rep = 0; rep < 2; ++rep) {
      int row = r + rep * 64;
      v8h q = *(const v8h*)(Qh + (size_t)(m_base + row) * DIM + k0 + part * 8);
      *(v8h*)(&bufA[row * 32 + part * 8]) = q * g;             // packed f16 mul
    }
  };

  auto compute = [&](const _Float16* bufA, const _Float16* bufB) {
    V16 af[2];
#pragma unroll
    for (int mt = 0; mt < 2; ++mt) {
      int row = wm * 32 + mt * 16 + rr;
      af[mt].h[0] = *(const v8h*)(&bufA[row * 32 + klo * 2]);
      af[mt].h[1] = *(const v8h*)(&bufA[row * 32 + klo * 2 + 8]);
    }
#pragma unroll
    for (int nt = 0; nt < 4; ++nt) {
      int col = wn * 64 + nt * 16 + rr;
      v16h bf = *(const v16h*)(&bufB[col * 32 + hi * 16]);
#pragma unroll
      for (int mt = 0; mt < 2; ++mt) {
        acc[mt][nt] = __builtin_amdgcn_wmma_f32_16x16x32_f16(
            false, af[mt].v, false, bf, (short)0, acc[mt][nt], false, false);
      }
    }
  };

  stage(0, ldsA, ldsB);
#pragma unroll 1
  for (int kc = 0; kc < 32; kc += 2) {
    wait_async0();
    __syncthreads();
    stage(kc + 1, ldsA + G2_T, ldsB + G2_T);   // kc+1 <= 31 always
    compute(ldsA, ldsB);

    wait_async0();
    __syncthreads();
    if (kc + 2 < 32) stage(kc + 2, ldsA, ldsB);
    compute(ldsA + G2_T, ldsB + G2_T);
  }

  // epilogue: + bo + X residual, f32 out
#pragma unroll
  for (int mt = 0; mt < 2; ++mt) {
#pragma unroll
    for (int nt = 0; nt < 4; ++nt) {
      int n = n_base + wn * 64 + nt * 16 + rr;
      float b0 = bo[n];
      int m0 = m_base + wm * 32 + mt * 16 + hi * 8;
#pragma unroll
      for (int j = 0; j < 8; ++j) {
        size_t idx = (size_t)(m0 + j) * DIM + n;
        out[idx] = acc[mt][nt][j] + b0 + X[idx];
      }
    }
  }
}

// ---------------- launch ----------------
extern "C" void kernel_launch(void* const* d_in, const int* in_sizes, int n_in,
                              void* d_out, int out_size, void* d_ws, size_t ws_size,
                              hipStream_t stream) {
  const float* X  = (const float*)d_in[0];
  const float* Wi = (const float*)d_in[1];
  const float* bi = (const float*)d_in[2];
  const float* Wo = (const float*)d_in[3];
  const float* bo = (const float*)d_in[4];
  float* out = (float*)d_out;

  char* ws = (char*)d_ws;
  _Float16* Xh   = (_Float16*)(ws);                 // 67,108,864 B
  _Float16* WiT  = (_Float16*)(ws +  67108864);     //  6,291,456 B  [3072][1024]
  _Float16* WoT  = (_Float16*)(ws +  73400320);     //  2,097,152 B  [1024][1024]
  _Float16* Qh   = (_Float16*)(ws +  75497472);     // 67,108,864 B
  float*    Aacc = (float*)   (ws + 142606336);     //     16,384 B
  _Float16* Ah   = (_Float16*)(ws + 142622720);     //      8,192 B

  hipMemsetAsync(Aacc, 0, 4 * DIM * sizeof(float), stream);

  cvt_x<<<(MTOT * DIM / 4) / 256, 256, 0, stream>>>(X, Xh, MTOT * DIM / 4);

  dim3 tb(32, 8);
  transpose_cvt<<<dim3(3 * DIM / 32, DIM / 32), tb, 0, stream>>>(Wi, WiT, DIM, 3 * DIM);
  transpose_cvt<<<dim3(DIM / 32, DIM / 32), tb, 0, stream>>>(Wo, WoT, DIM, DIM);

  gemm1_qkv<<<dim3(DIM / 64, MTOT / 128), 256, 0, stream>>>(Xh, WiT, bi, Qh, Aacc);

  finalize_gate<<<(4 * DIM) / 256, 256, 0, stream>>>(Aacc, Ah, 4 * DIM);

  gemm2_out<<<dim3(DIM / 128, MTOT / 128), 256, 0, stream>>>(Qh, Ah, WoT, bo, X, out);
}
